// SelfAttention_6889127543338
// MI455X (gfx1250) — matrix operand order repaired and checked by reading, not compiled, not run on hardware
//
#include <hip/hip_runtime.h>
#include <hip/hip_bf16.h>

#define CDIM 256
#define NDIM 4096
#define BDIM 4
#define DPAD 32   // qk depth padded 16 -> 32 (zero-filled) so one K=32 WMMA covers it
#define OROWS (2 * 16 + CDIM)   // 288 packed output rows: q(16) | k(16) | v(256)

typedef __attribute__((ext_vector_type(16))) __bf16 v16bf;
typedef __attribute__((ext_vector_type(8)))  float  v8f;

union ABOp {               // one WMMA 16-bit A/B operand: 8 VGPRs = 16 bf16 per lane
  v16bf v;
  unsigned short u[16];
  uint4 q[2];
};

__device__ __forceinline__ unsigned short f2bf(float f) {
  union { __bf16 h; unsigned short u; } c;
  c.h = (__bf16)f;            // native gfx1250 f32->bf16 convert
  return c.u;
}

// ---------------------------------------------------------------------------
// Kernel 0: prep.
//   blocks [0, B*64*4): tiled 64x64 transpose+convert  x[b][c][n] -> xT[b][n][c] bf16
//   blocks >= that    : pack Wcat bf16 [288][256] = [wq;wk;wv]
// ---------------------------------------------------------------------------
#define TRTILES (BDIM * (NDIM / 64) * (CDIM / 64))   // 4*64*4 = 1024
#define WBLOCKS 8

__global__ __launch_bounds__(256)
void prep_kernel(const float* __restrict__ x,
                 const float* __restrict__ wq, const float* __restrict__ wk,
                 const float* __restrict__ wvm,
                 unsigned short* __restrict__ xT,
                 unsigned short* __restrict__ Wc) {
  const int tid = threadIdx.x;
  if (blockIdx.x < TRTILES) {
    __shared__ unsigned short t[64][65];            // padded to dodge bank conflicts
    int tile = blockIdx.x;
    int b  = tile >> 8;                              // 256 tiles per batch
    int r  = tile & 255;
    int n0 = (r & 63) * 64;
    int c0 = (r >> 6) * 64;
    const float* xb = x + (size_t)b * CDIM * NDIM;
#pragma unroll
    for (int k = 0; k < 16; ++k) {                   // read coalesced in n
      int idx = tid + k * 256;
      int i = idx >> 6, j = idx & 63;                // i: c offset, j: n offset
      t[i][j] = f2bf(xb[(size_t)(c0 + i) * NDIM + n0 + j]);
    }
    __syncthreads();
    unsigned short* xo = xT + ((size_t)b * NDIM) * CDIM;
#pragma unroll
    for (int k = 0; k < 16; ++k) {                   // write coalesced in c
      int idx = tid + k * 256;
      int i = idx >> 6, j = idx & 63;                // i: n offset, j: c offset
      xo[(size_t)(n0 + i) * CDIM + c0 + j] = t[j][i];
    }
  } else {
    int base = (blockIdx.x - TRTILES) * 256 + tid;
    for (int idx = base; idx < OROWS * CDIM; idx += WBLOCKS * 256) {
      int o = idx >> 8, c = idx & 255;
      float v;
      if (o < 16)       v = wq[o * CDIM + c];
      else if (o < 32)  v = wk[(o - 16) * CDIM + c];
      else              v = wvm[(o - 32) * CDIM + c];
      Wc[idx] = f2bf(v);
    }
  }
}

// ---------------------------------------------------------------------------
// Kernel 1: fused 1x1-conv projections  Y = W x + b  (pure WMMA bf16 GEMM).
//   blockIdx = (n-chunk of 64, o-tile 0..17, batch). 128 threads = 4 waves,
//   wave w owns n-subtile w. No LDS, no barriers: both operands are direct
//   16B global b128 loads at k offsets kb and 16+kb (A from Wcat, B from xT).
//   o-tile 0 -> Q[n][32] (zero-padded), 1 -> K[n][32], 2..17 -> V[c][n].
// ---------------------------------------------------------------------------
__global__ __launch_bounds__(128)
void proj_kernel(const unsigned short* __restrict__ xT,
                 const unsigned short* __restrict__ Wc,
                 const float* __restrict__ bq, const float* __restrict__ bk,
                 const float* __restrict__ bv,
                 unsigned short* __restrict__ Qd,
                 unsigned short* __restrict__ Kd,
                 unsigned short* __restrict__ Vh) {
  const int tid  = threadIdx.x;
  const int lane = tid & 31;
  const int wid  = tid >> 5;           // 0..3: n-subtile
  const int hf   = lane >> 4;          // lane half selects K-chunks per ISA layout
  const int l15  = lane & 15;
  const int kb   = hf * 8;
  const int n0   = blockIdx.x * 64;
  const int ot   = blockIdx.y;         // 0..17
  const int b    = blockIdx.z;

  const unsigned short* wr = Wc + (size_t)(ot * 16 + l15) * CDIM;            // A row M=l15
  const unsigned short* xr = xT + ((size_t)b * NDIM + n0 + wid * 16 + l15) * CDIM; // B col N=l15

  v8f acc = {};
#pragma unroll
  for (int ci = 0; ci < 8; ++ci) {     // K loop over c in steps of 32
    const int cb = ci * 32;
    ABOp A, Bm;
    A.q[0]  = *(const uint4*)(wr + cb + kb);
    A.q[1]  = *(const uint4*)(wr + cb + 16 + kb);
    Bm.q[0] = *(const uint4*)(xr + cb + kb);
    Bm.q[1] = *(const uint4*)(xr + cb + 16 + kb);
    acc = __builtin_amdgcn_wmma_f32_16x16x32_bf16(false, A.v, false, Bm.v,
                                                  (short)0, acc, false, false);
  }

  const float* bsel = (ot == 0) ? bq : (ot == 1) ? bk : bv;
  const int    ob   = (ot < 2) ? 0 : (ot - 2) * 16;
  const int    n    = n0 + wid * 16 + l15;     // D col N = l15
#pragma unroll
  for (int g = 0; g < 8; ++g) {                // D row M = g + 8*hf
    int ol = g + 8 * hf;
    unsigned short hv = f2bf(acc[g] + bsel[ob + ol]);
    if (ot == 0) {
      size_t r = (((size_t)b * NDIM) + n) * DPAD;
      Qd[r + ol] = hv;  Qd[r + 16 + ol] = 0;   // zero pad d=16..31
    } else if (ot == 1) {
      size_t r = (((size_t)b * NDIM) + n) * DPAD;
      Kd[r + ol] = hv;  Kd[r + 16 + ol] = 0;
    } else {
      Vh[(((size_t)b * CDIM) + ob + ol) * NDIM + n] = hv;
    }
  }
}

// ---------------------------------------------------------------------------
// Kernel 2: streaming (two-pass flash) attention + gamma residual.
//   block = (32 queries, 1 batch), 256 threads = 8 waves.
//   wave w: S-tile (qt=w&1, kt=w>>1); O-tiles (qt=w&1, c-tiles (w>>1)*4..+3).
//   S-gemm:  D[key,q] = Key[16k x 32d] * Q^T[32d x 16q]  -> per-query stats in-lane.
//   O-gemm:  D[c,q]  += V[16c x 32m]  * P^T[32m x 16q]   (P staged in LDS as bf16).
// ---------------------------------------------------------------------------
__global__ __launch_bounds__(256)
void attn_kernel(const float* __restrict__ x,
                 const unsigned short* __restrict__ Qd,
                 const unsigned short* __restrict__ Kd,
                 const unsigned short* __restrict__ Vh,
                 const float* __restrict__ gamma,
                 float* __restrict__ out) {
  __shared__ __align__(16) unsigned short ldsP[2][16][72];  // [qt][q][m(64)+pad]
  __shared__ float wm[8][16], wsum[8][16];
  __shared__ float statm[2][16], statrs[2][16];

  const int tid  = threadIdx.x;
  const int lane = tid & 31;
  const int w    = tid >> 5;
  const int hf   = lane >> 4;
  const int l15  = lane & 15;
  const int kb   = hf * 8;
  const int b    = blockIdx.y;
  const int n0   = blockIdx.x * 32;

  const int qt    = w & 1;
  const int kt    = w >> 1;       // 0..3: key subtile within 64-key chunk
  const int cbase = (w >> 1) * 64;

  const unsigned short* Kb = Kd + ((size_t)b * NDIM) * DPAD;
  const unsigned short* Vb = Vh + ((size_t)b * CDIM) * NDIM;

  // B operand (Q^T) invariant across the key loop: lane = query column, d in table order (8 at kb, 8 at 16+kb)
  ABOp Qop;
  {
    const unsigned short* qp =
        Qd + (((size_t)b * NDIM) + n0 + qt * 16 + l15) * DPAD + kb;
    Qop.q[0] = *(const uint4*)(qp);
    Qop.q[1] = *(const uint4*)(qp + 16);
  }

  // ---- Pass 1: per-query online (max, sum-exp) over all 4096 keys ----
  float m_loc = -INFINITY, s_loc = 0.0f;
  for (int ch = 0; ch < NDIM / 64; ++ch) {
    const unsigned short* kp = Kb + (size_t)(ch * 64 + kt * 16 + l15) * DPAD;
    ABOp Ak;                                    // A = Key tile, row M = key = l15
    Ak.q[0] = *(const uint4*)(kp + kb);
    Ak.q[1] = *(const uint4*)(kp + 16 + kb);
    v8f cz = {};
    v8f s = __builtin_amdgcn_wmma_f32_16x16x32_bf16(false, Ak.v, false, Qop.v,
                                                    (short)0, cz, false, false);
    float tm = s[0];
#pragma unroll
    for (int g = 1; g < 8; ++g) tm = fmaxf(tm, s[g]);
    float mnew = fmaxf(m_loc, tm);
    float add = 0.0f;
#pragma unroll
    for (int g = 0; g < 8; ++g) add += __expf(s[g] - mnew);
    s_loc = s_loc * __expf(m_loc - mnew) + add;
    m_loc = mnew;
  }
  { // merge key-halves held by lane pair (l, l^16) — same query column
    float mo = __shfl_xor(m_loc, 16, 32);
    float so = __shfl_xor(s_loc, 16, 32);
    float mn = fmaxf(m_loc, mo);
    s_loc = s_loc * __expf(m_loc - mn) + so * __expf(mo - mn);
    m_loc = mn;
  }
  if (lane < 16) { wm[w][l15] = m_loc; wsum[w][l15] = s_loc; }
  __syncthreads();
  if (tid < 32) {  // merge the 4 waves sharing each query subtile
    int tq = tid >> 4, q = tid & 15;
    float m = -INFINITY, ss = 0.0f;
#pragma unroll
    for (int ww = 0; ww < 4; ++ww) {
      float mi = wm[tq + ww * 2][q], si = wsum[tq + ww * 2][q];
      float mn = fmaxf(m, mi);
      ss = ss * __expf(m - mn) + si * __expf(mi - mn);
      m = mn;
    }
    statm[tq][q] = m;
    statrs[tq][q] = 1.0f / ss;
  }
  __syncthreads();

  const float mq = statm[qt][l15];
  const float rs = statrs[qt][l15];
  const float g0 = gamma[0];

  v8f acc[4] = {{}, {}, {}, {}};

  // ---- Pass 2: recompute S, P = exp(S - m), O += V * P^T ----
  for (int ch = 0; ch < NDIM / 64; ++ch) {
    const unsigned short* kp = Kb + (size_t)(ch * 64 + kt * 16 + l15) * DPAD;
    ABOp Ak;
    Ak.q[0] = *(const uint4*)(kp + kb);
    Ak.q[1] = *(const uint4*)(kp + 16 + kb);
    v8f cz = {};
    v8f s = __builtin_amdgcn_wmma_f32_16x16x32_bf16(false, Ak.v, false, Qop.v,
                                                    (short)0, cz, false, false);
    union { uint4 q; unsigned short u[8]; } pk;   // 8 probs are LDS-contiguous
#pragma unroll
    for (int g = 0; g < 8; ++g) pk.u[g] = f2bf(__expf(s[g] - mq));
    *(uint4*)&ldsP[qt][l15][kt * 16 + 8 * hf] = pk.q;
    __syncthreads();

#pragma unroll
    for (int sc = 0; sc < 2; ++sc) {              // 32-key subchunks
      ABOp Bp;                                    // B = P^T: lane = query col, m in table order
      const unsigned short* pp = &ldsP[qt][l15][sc * 32 + kb];
      Bp.q[0] = *(const uint4*)(pp);
      Bp.q[1] = *(const uint4*)(pp + 16);
      const int mb = ch * 64 + sc * 32;
#pragma unroll
      for (int j = 0; j < 4; ++j) {               // A = V tile [16c x 32m]
        const unsigned short* vp = Vb + (size_t)(cbase + j * 16 + l15) * NDIM + mb;
        ABOp Av;
        Av.q[0] = *(const uint4*)(vp + kb);
        Av.q[1] = *(const uint4*)(vp + 16 + kb);
        acc[j] = __builtin_amdgcn_wmma_f32_16x16x32_bf16(false, Av.v, false, Bp.v,
                                                         (short)0, acc[j], false, false);
      }
    }
    __syncthreads();   // protect ldsP before next chunk's writes
  }

  // ---- epilogue: out = gamma * (O / sum) + x ----
  const int n = n0 + qt * 16 + l15;
#pragma unroll
  for (int j = 0; j < 4; ++j) {
#pragma unroll
    for (int g = 0; g < 8; ++g) {
      int c = cbase + j * 16 + g + 8 * hf;
      size_t idx = (((size_t)b * CDIM) + c) * NDIM + n;
      out[idx] = g0 * (acc[j][g] * rs) + x[idx];
    }
  }
}

// ---------------------------------------------------------------------------
extern "C" void kernel_launch(void* const* d_in, const int* in_sizes, int n_in,
                              void* d_out, int out_size, void* d_ws, size_t ws_size,
                              hipStream_t stream) {
  const float* x  = (const float*)d_in[0];
  const float* wq = (const float*)d_in[1];
  const float* bq = (const float*)d_in[2];
  const float* wk = (const float*)d_in[3];
  const float* bk = (const float*)d_in[4];
  const float* wv = (const float*)d_in[5];
  const float* bv = (const float*)d_in[6];
  const float* gm = (const float*)d_in[7];
  float* out = (float*)d_out;

  // workspace: Q/K [B][N][32] + V [B][C][N] + xT [B][N][C] + Wcat [288][256], all bf16
  char* ws = (char*)d_ws;
  const size_t qk_bytes = (size_t)BDIM * NDIM * DPAD * sizeof(unsigned short);
  const size_t cv_bytes = (size_t)BDIM * CDIM * NDIM * sizeof(unsigned short);
  unsigned short* Qd = (unsigned short*)ws;
  unsigned short* Kd = (unsigned short*)(ws + qk_bytes);
  unsigned short* Vh = (unsigned short*)(ws + 2 * qk_bytes);
  unsigned short* xT = (unsigned short*)(ws + 2 * qk_bytes + cv_bytes);
  unsigned short* Wc = (unsigned short*)(ws + 2 * qk_bytes + 2 * cv_bytes);

  prep_kernel<<<TRTILES + WBLOCKS, 256, 0, stream>>>(x, wq, wk, wv, xT, Wc);
  proj_kernel<<<dim3(NDIM / 64, 2 + CDIM / 16, BDIM), 128, 0, stream>>>(
      xT, Wc, bq, bk, bv, Qd, Kd, Vh);
  attn_kernel<<<dim3(NDIM / 32, BDIM), 256, 0, stream>>>(
      x, Qd, Kd, Vh, gm, out);
}
